// Encoder_69672959475929
// MI455X (gfx1250) — matrix operand order repaired and checked by reading, not compiled, not run on hardware
//
#include <hip/hip_runtime.h>
#include <hip/hip_bf16.h>
#include <math.h>

#define B_   8
#define L_   512
#define D_   1024
#define H_   16
#define DFF_ 4096
#define NL_  6
#define DH_  64
#define EPS_ 1e-5f

typedef __attribute__((ext_vector_type(16))) __bf16 v16bf;
typedef __attribute__((ext_vector_type(8)))  float  v8f;
typedef __attribute__((ext_vector_type(4)))  unsigned v4u;
typedef __attribute__((ext_vector_type(4)))  int      v4i;
typedef __attribute__((ext_vector_type(8)))  int      v8i;

__device__ __forceinline__ __bf16 f2bf(float f) {
    unsigned u = __builtin_bit_cast(unsigned, f);
    unsigned r = u + 0x7FFFu + ((u >> 16) & 1u);   // round-to-nearest-even
    unsigned short h = (unsigned short)(r >> 16);
    return __builtin_bit_cast(__bf16, h);
}

// ---------------------------------------------------------------------------
// TDM: async 2D tile load (tileRows x 64 bf16 elems) global -> LDS.
// D# per CDNA5 ISA: group0 {count=1, lds_addr, global_addr,
// type=2}; group1 {data_size=2B, pad_enable, pad_interval=32dw (code 4),
// pad_amount=4dw (code 3), tile 64 x tileRows, dim0 stride = K elems}.
// LDS row stride = 32dw + 4dw pad = 72 bf16 elems (matches LDS tiles).
// Issued by wave 0 only; tracked with TENSORcnt (in-order per wave).
// ---------------------------------------------------------------------------
__device__ __forceinline__ void tdm_load_tile(unsigned lds_off, const __bf16* gptr,
                                              int K, int tileRows) {
    unsigned long long ga = (unsigned long long)(uintptr_t)gptr;
    v4u g0;
    g0[0] = 1u;                                           // count=1, user mode
    g0[1] = lds_off;                                      // LDS byte address
    g0[2] = (unsigned)ga;                                 // global_addr[31:0]
    g0[3] = (unsigned)((ga >> 32) & 0x01FFFFFFu) | (2u << 30);  // addr hi | type=2
    v8i g1;
    unsigned td0 = (unsigned)K;        // tensor_dim0 (elements)
    unsigned td1 = 1u << 20;           // tensor_dim1 (large; tiles in-bounds)
    g1[0] = (int)((1u << 16)           // data_size = 1 -> 2 bytes
                | (1u << 20)           // pad_enable
                | (4u << 22)           // pad_interval = 32 dwords
                | (3u << 25));         // pad_amount   = 4 dwords
    g1[1] = (int)((td0 & 0xFFFFu) << 16);                 // dim0[15:0] @ [31:16]
    g1[2] = (int)((td0 >> 16) | ((td1 & 0xFFFFu) << 16)); // dim0 hi | dim1 lo
    g1[3] = (int)((td1 >> 16) | (64u << 16));             // dim1 hi | tile_dim0=64
    g1[4] = (int)(unsigned)tileRows;                      // tile_dim1
    g1[5] = (int)(unsigned)K;                             // tensor_dim0_stride lo
    g1[6] = 0;
    g1[7] = 0;
    v4i gz = {0, 0, 0, 0};
#if __clang_major__ >= 23
    v8i gz8 = {0, 0, 0, 0, 0, 0, 0, 0};
    __builtin_amdgcn_tensor_load_to_lds(g0, g1, gz, gz, gz8, 0);
#else
    __builtin_amdgcn_tensor_load_to_lds(g0, g1, gz, gz, 0);
#endif
}

// ---------------------------------------------------------------------------
// Tiled fp32[K,N] -> bf16[N,K] transpose-convert (weights, once per GEMM).
// ---------------------------------------------------------------------------
__global__ __launch_bounds__(256)
void transpose_f2bf_kernel(const float* __restrict__ src, __bf16* __restrict__ dst,
                           int K, int N) {
    __shared__ float tile[32][33];
    int k0 = blockIdx.x * 32, n0 = blockIdx.y * 32;
    int tx = threadIdx.x & 31, ty = threadIdx.x >> 5;     // 8 rows / pass
#pragma unroll
    for (int r = 0; r < 32; r += 8)
        tile[ty + r][tx] = src[(long long)(k0 + ty + r) * N + n0 + tx];
    __syncthreads();
#pragma unroll
    for (int r = 0; r < 32; r += 8)
        dst[(long long)(n0 + ty + r) * K + k0 + tx] = f2bf(tile[tx][ty + r]);
}

// ---------------------------------------------------------------------------
// Embedding: x = tok_emb[seq] + PE + seg_emb[seg]   (fp32 + bf16 copies)
// ---------------------------------------------------------------------------
__global__ __launch_bounds__(256)
void embed_kernel(const int* __restrict__ seq, const int* __restrict__ seg,
                  const float* __restrict__ tok, const float* __restrict__ se,
                  float* __restrict__ xf, __bf16* __restrict__ xb) {
    int token = blockIdx.x;                 // b*L + l
    int l = token & (L_ - 1);
    int s = seq[token];
    int g = seg[token];
    int t = threadIdx.x;
#pragma unroll
    for (int e = 0; e < 4; ++e) {
        int d = t * 4 + e;
        int i = d >> 1;
        float div  = __expf(-9.210340371976184f * (float)(2 * i) / (float)D_);
        float ang  = (float)l * div;
        float pe   = (d & 1) ? cosf(ang) : sinf(ang);
        float val  = tok[(long long)s * D_ + d] + pe + se[(long long)g * D_ + d];
        long long idx = (long long)token * D_ + d;
        xf[idx] = val;
        xb[idx] = f2bf(val);
    }
}

// ---------------------------------------------------------------------------
// bf16 WMMA GEMM, TDM double-buffered, 64-wide K stages.
//   C[z] = act(A[z][M,K] @ Wt[z][N,K]^T + bias)      (Wt is K-major)
// Block tile 128(M) x BN(N), BN = NFRAG*32. 8 waves (4x2); each wave computes
// 32 x (BN/2) via 2 x NFRAG 16x16 frags, 2 K-substeps -> 4*NFRAG WMMAs/stage.
// act: 0=none, 1=exact GELU.  storeMode: 0=row-major C, 1=V-transpose scatter.
// ---------------------------------------------------------------------------
template <int NFRAG>
__global__ __launch_bounds__(256)
void gemm_bf16_kernel(const __bf16* __restrict__ A, const __bf16* __restrict__ Wt,
                      const float* __restrict__ bias,
                      float* __restrict__ outF, __bf16* __restrict__ outB,
                      int M, int N, int K, int act, int storeMode,
                      long long strideA, long long strideW, long long strideC) {
    constexpr int BN = NFRAG * 32;
    __shared__ __align__(16) __bf16 As[2][128][72];   // [buf][m][k] padded
    __shared__ __align__(16) __bf16 Bs[2][BN][72];    // [buf][n][k] padded

    const int tid  = threadIdx.x;
    const int lane = tid & 31;
    const int w    = tid >> 5;
    const int waveM = w & 3;          // 32-row slab
    const int waveN = w >> 2;         // (BN/2)-col slab
    const int half  = lane >> 4;
    const int l16   = lane & 15;

    const long long z = blockIdx.z;
    const int mBase = blockIdx.x * 128;
    const int nBase = blockIdx.y * BN;
    const __bf16* Ab = A  + z * strideA + (long long)mBase * K;
    const __bf16* Wb = Wt + z * strideW + (long long)nBase * K;

    unsigned asOff[2], bsOff[2];
    asOff[0] = (unsigned)(unsigned long long)(uintptr_t)&As[0][0][0];
    asOff[1] = (unsigned)(unsigned long long)(uintptr_t)&As[1][0][0];
    bsOff[0] = (unsigned)(unsigned long long)(uintptr_t)&Bs[0][0][0];
    bsOff[1] = (unsigned)(unsigned long long)(uintptr_t)&Bs[1][0][0];

    v8f acc[2][NFRAG] = {};
    const int nk = K >> 6;                          // 64-wide K stages

    if (w == 0) {                                   // wave 0 drives the TDM
        tdm_load_tile(asOff[0], Ab, K, 128);
        tdm_load_tile(bsOff[0], Wb, K, BN);
    }

    for (int kt = 0; kt < nk; ++kt) {
        const int cur = kt & 1;
        if (w == 0) {
            if (kt + 1 < nk) {
                tdm_load_tile(asOff[cur ^ 1], Ab + (kt + 1) * 64, K, 128);
                tdm_load_tile(bsOff[cur ^ 1], Wb + (kt + 1) * 64, K, BN);
                __builtin_amdgcn_s_wait_tensorcnt(2);  // in-order: stage kt done
            } else {
                __builtin_amdgcn_s_wait_tensorcnt(0);
            }
        }
        __syncthreads();                            // stage kt visible to all waves

#pragma unroll
        for (int ks = 0; ks < 2; ++ks) {            // two 32-wide K substeps
            const int kb = ks * 32;
            v16bf afr[2], bfr[NFRAG];
#pragma unroll
            for (int mi = 0; mi < 2; ++mi) {
                int r = waveM * 32 + mi * 16 + l16;
#pragma unroll
                for (int j = 0; j < 8; ++j) {
                    afr[mi][j]     = As[cur][r][kb + half * 8 + j];
                    afr[mi][j + 8] = As[cur][r][kb + 16 + half * 8 + j];
                }
            }
#pragma unroll
            for (int ni = 0; ni < NFRAG; ++ni) {
                int c = waveN * (BN / 2) + ni * 16 + l16;
#pragma unroll
                for (int j = 0; j < 8; ++j) {
                    bfr[ni][j]     = Bs[cur][c][kb + half * 8 + j];
                    bfr[ni][j + 8] = Bs[cur][c][kb + 16 + half * 8 + j];
                }
            }
#pragma unroll
            for (int mi = 0; mi < 2; ++mi)
#pragma unroll
                for (int ni = 0; ni < NFRAG; ++ni)
                    acc[mi][ni] = __builtin_amdgcn_wmma_f32_16x16x32_bf16(
                        false, afr[mi], false, bfr[ni], (short)0, acc[mi][ni],
                        false, false);
        }
        __syncthreads();                            // done with buf before overwrite
    }

    // ---- epilogue ----
#pragma unroll
    for (int mi = 0; mi < 2; ++mi) {
        int gm0 = mBase + waveM * 32 + mi * 16 + half * 8;
#pragma unroll
        for (int ni = 0; ni < NFRAG; ++ni) {
            int gn = nBase + waveN * (BN / 2) + ni * 16 + l16;
            float bv = bias ? bias[gn] : 0.0f;
#pragma unroll
            for (int v = 0; v < 8; ++v) {
                float val = acc[mi][ni][v] + bv;
                if (act == 1)
                    val = 0.5f * val * (1.0f + erff(val * 0.7071067811865475f));
                if (storeMode == 0) {
                    long long idx = z * strideC + (long long)(gm0 + v) * N + gn;
                    if (outF) outF[idx] = val;
                    if (outB) outB[idx] = f2bf(val);
                } else {
                    // V^T scatter: token row -> per-(b,h) [DH, L] layout
                    int token = gm0 + v;            // b*L + l2
                    int bb  = token >> 9;
                    int l2  = token & (L_ - 1);
                    int f   = (l2 << 10) + gn;      // flat within [L*D)
                    int h   = f >> 15;              // / (L*DH)
                    int rem = f & 32767;
                    int lpos = rem >> 6;
                    int d    = rem & 63;
                    long long idx = ((((long long)bb * H_ + h) * DH_) + d) * L_ + lpos;
                    outB[idx] = f2bf(val);
                }
            }
        }
    }
}

// ---------------------------------------------------------------------------
// Attention scores:  S[h] = (Q[h] @ K[h]^T) * scale      per batch b
// Q,K contiguous [L,64] per head, both K-contiguous -> direct global frags.
// ---------------------------------------------------------------------------
__global__ __launch_bounds__(256)
void attn_scores_kernel(const __bf16* __restrict__ q, const __bf16* __restrict__ k,
                        float* __restrict__ scores, float scale) {
    int h = blockIdx.z;
    const __bf16* Q  = q + (long long)h * L_ * DH_;
    const __bf16* Km = k + (long long)h * L_ * DH_;
    float* S = scores + (long long)h * L_ * L_;

    int tid = threadIdx.x, lane = tid & 31, w = tid >> 5;
    int half = lane >> 4, l16 = lane & 15;
    int mTile = blockIdx.x * 16;
    int nTile = blockIdx.y * 128 + w * 16;

    int row = mTile + l16;
    int col = nTile + l16;
    v8f acc = {};
#pragma unroll
    for (int k0 = 0; k0 < DH_; k0 += 32) {
        v16bf a, b;
        const __bf16* qr = Q + (long long)row * DH_ + k0;
#pragma unroll
        for (int j = 0; j < 8; ++j) {
            a[j]     = qr[half * 8 + j];
            a[j + 8] = qr[16 + half * 8 + j];
        }
        const __bf16* kr = Km + (long long)col * DH_ + k0;
#pragma unroll
        for (int j = 0; j < 8; ++j) {
            b[j]     = kr[half * 8 + j];
            b[j + 8] = kr[16 + half * 8 + j];
        }
        acc = __builtin_amdgcn_wmma_f32_16x16x32_bf16(
            false, a, false, b, (short)0, acc, false, false);
    }
    int gm = mTile + half * 8;
#pragma unroll
    for (int v = 0; v < 8; ++v)
        S[(long long)(gm + v) * L_ + nTile + l16] = acc[v] * scale;
}

// ---------------------------------------------------------------------------
// Row softmax with key-padding mask (cols where seq[col]==0 get -1e9).
// ---------------------------------------------------------------------------
__global__ __launch_bounds__(256)
void softmax_kernel(const float* __restrict__ scores, const int* __restrict__ seq,
                    __bf16* __restrict__ attn) {
    long long rowBase = (long long)blockIdx.x * L_;
    int t = threadIdx.x;
    __shared__ float red[256];

    float v0 = scores[rowBase + t];
    float v1 = scores[rowBase + t + 256];
    if (seq[t]       == 0) v0 = -1.0e9f;
    if (seq[t + 256] == 0) v1 = -1.0e9f;

    red[t] = fmaxf(v0, v1);
    __syncthreads();
    for (int s = 128; s > 0; s >>= 1) {
        if (t < s) red[t] = fmaxf(red[t], red[t + s]);
        __syncthreads();
    }
    float mx = red[0];
    __syncthreads();

    float e0 = __expf(v0 - mx), e1 = __expf(v1 - mx);
    red[t] = e0 + e1;
    __syncthreads();
    for (int s = 128; s > 0; s >>= 1) {
        if (t < s) red[t] += red[t + s];
        __syncthreads();
    }
    float inv = 1.0f / red[0];
    attn[rowBase + t]       = f2bf(e0 * inv);
    attn[rowBase + t + 256] = f2bf(e1 * inv);
}

// ---------------------------------------------------------------------------
// Fused residual add + LayerNorm; one block per row of D=1024.
// ---------------------------------------------------------------------------
__global__ __launch_bounds__(256)
void add_ln_kernel(const float* __restrict__ a, const float* __restrict__ b,
                   float* __restrict__ dstF, __bf16* __restrict__ dstB) {
    long long row = blockIdx.x;
    const float* ar = a + row * D_;
    const float* br = b + row * D_;
    int t = threadIdx.x;

    float x[4], s = 0.f, s2 = 0.f;
#pragma unroll
    for (int e = 0; e < 4; ++e) {
        int d = t + e * 256;
        float v = ar[d] + br[d];
        x[e] = v; s += v; s2 += v * v;
    }
    __shared__ float r1[256], r2[256];
    r1[t] = s; r2[t] = s2;
    __syncthreads();
    for (int st = 128; st > 0; st >>= 1) {
        if (t < st) { r1[t] += r1[t + st]; r2[t] += r2[t + st]; }
        __syncthreads();
    }
    float mean = r1[0] * (1.0f / D_);
    float var  = r2[0] * (1.0f / D_) - mean * mean;
    float rstd = rsqrtf(var + EPS_);
#pragma unroll
    for (int e = 0; e < 4; ++e) {
        int d = t + e * 256;
        float v = (x[e] - mean) * rstd;
        dstF[row * D_ + d] = v;
        dstB[row * D_ + d] = f2bf(v);
    }
}

// ---------------------------------------------------------------------------
// Host orchestration
// ---------------------------------------------------------------------------
extern "C" void kernel_launch(void* const* d_in, const int* in_sizes, int n_in,
                              void* d_out, int out_size, void* d_ws, size_t ws_size,
                              hipStream_t stream) {
    const int*   seq  = (const int*)d_in[0];
    const int*   seg  = (const int*)d_in[1];
    const float* tok  = (const float*)d_in[2];
    const float* sege = (const float*)d_in[3];
    const float* Wq = (const float*)d_in[4];  const float* bq = (const float*)d_in[5];
    const float* Wk = (const float*)d_in[6];  const float* bk = (const float*)d_in[7];
    const float* Wv = (const float*)d_in[8];  const float* bv = (const float*)d_in[9];
    const float* Wo = (const float*)d_in[10]; const float* bo = (const float*)d_in[11];
    const float* W1 = (const float*)d_in[12]; const float* b1 = (const float*)d_in[13];
    const float* W2 = (const float*)d_in[14]; const float* b2 = (const float*)d_in[15];

    const long long BLD = (long long)B_ * L_ * D_;
    char* ws = (char*)d_ws;
    size_t off = 0;
    auto alloc = [&](size_t bytes) -> void* {
        void* p = ws + off;
        off += (bytes + 255) & ~(size_t)255;
        return p;
    };
    float*  x_f   = (float*)alloc(BLD * 4);
    __bf16* x_b   = (__bf16*)alloc(BLD * 2);
    __bf16* qb    = (__bf16*)alloc(BLD * 2);
    __bf16* kb    = (__bf16*)alloc(BLD * 2);
    __bf16* vt    = (__bf16*)alloc(BLD * 2);   // V^T per (b,h): [DH, L]
    __bf16* ctx_b = (__bf16*)alloc(BLD * 2);
    __bf16* h_b   = (__bf16*)alloc((long long)B_ * L_ * DFF_ * 2);
    float*  tmp_f = (float*)alloc(BLD * 4);
    float*  sc_f  = (float*)alloc((long long)H_ * L_ * L_ * 4);
    __bf16* at_b  = (__bf16*)alloc((long long)H_ * L_ * L_ * 2);
    __bf16* wt    = (__bf16*)alloc((long long)D_ * DFF_ * 2);  // transposed weights

    const int Mtok = B_ * L_;

    embed_kernel<<<Mtok, 256, 0, stream>>>(seq, seg, tok, sege, x_f, x_b);

    for (int i = 0; i < NL_; ++i) {
        const float* Wq_i = Wq + (size_t)i * D_ * D_;   const float* bq_i = bq + (size_t)i * D_;
        const float* Wk_i = Wk + (size_t)i * D_ * D_;   const float* bk_i = bk + (size_t)i * D_;
        const float* Wv_i = Wv + (size_t)i * D_ * D_;   const float* bv_i = bv + (size_t)i * D_;
        const float* Wo_i = Wo + (size_t)i * D_ * D_;   const float* bo_i = bo + (size_t)i * D_;
        const float* W1_i = W1 + (size_t)i * D_ * DFF_; const float* b1_i = b1 + (size_t)i * DFF_;
        const float* W2_i = W2 + (size_t)i * DFF_ * D_; const float* b2_i = b2 + (size_t)i * D_;

        dim3 gD(Mtok / 128, D_ / 128, 1);
        dim3 tDD(D_ / 32, D_ / 32, 1);

        // Q, K projections (bf16 out), V projection (V^T scatter)
        transpose_f2bf_kernel<<<tDD, 256, 0, stream>>>(Wq_i, wt, D_, D_);
        gemm_bf16_kernel<4><<<gD, 256, 0, stream>>>(x_b, wt, bq_i, nullptr, qb,
                                                    Mtok, D_, D_, 0, 0, 0, 0, 0);
        transpose_f2bf_kernel<<<tDD, 256, 0, stream>>>(Wk_i, wt, D_, D_);
        gemm_bf16_kernel<4><<<gD, 256, 0, stream>>>(x_b, wt, bk_i, nullptr, kb,
                                                    Mtok, D_, D_, 0, 0, 0, 0, 0);
        transpose_f2bf_kernel<<<tDD, 256, 0, stream>>>(Wv_i, wt, D_, D_);
        gemm_bf16_kernel<4><<<gD, 256, 0, stream>>>(x_b, wt, bv_i, nullptr, vt,
                                                    Mtok, D_, D_, 0, 1, 0, 0, 0);

        // attention per batch (scores buffer reused)
        for (int b = 0; b < B_; ++b) {
            const __bf16* qbb = qb + (long long)b * L_ * D_;
            const __bf16* kbb = kb + (long long)b * L_ * D_;
            const __bf16* vtb = vt + (long long)b * L_ * D_;   // b*H*DH*L
            attn_scores_kernel<<<dim3(L_ / 16, L_ / 128, H_), 256, 0, stream>>>(
                qbb, kbb, sc_f, 0.125f);
            softmax_kernel<<<H_ * L_, 256, 0, stream>>>(sc_f, seq + b * L_, at_b);
            // context = attn @ V, batched over heads via z (Wt = V^T per head)
            gemm_bf16_kernel<2><<<dim3(L_ / 128, 1, H_), 256, 0, stream>>>(
                at_b, vtb, nullptr, nullptr, ctx_b + (long long)b * L_ * D_,
                L_, DH_, L_, 0, 0,
                (long long)L_ * L_, (long long)DH_ * L_, (long long)L_ * DH_);
        }

        // output projection + add&norm
        transpose_f2bf_kernel<<<tDD, 256, 0, stream>>>(Wo_i, wt, D_, D_);
        gemm_bf16_kernel<4><<<gD, 256, 0, stream>>>(ctx_b, wt, bo_i, tmp_f, nullptr,
                                                    Mtok, D_, D_, 0, 0, 0, 0, 0);
        add_ln_kernel<<<Mtok, 256, 0, stream>>>(tmp_f, x_f, x_f, x_b);

        // FFN
        transpose_f2bf_kernel<<<dim3(D_ / 32, DFF_ / 32, 1), 256, 0, stream>>>(
            W1_i, wt, D_, DFF_);
        gemm_bf16_kernel<4><<<dim3(Mtok / 128, DFF_ / 128, 1), 256, 0, stream>>>(
            x_b, wt, b1_i, nullptr, h_b, Mtok, DFF_, D_, 1, 0, 0, 0, 0);
        transpose_f2bf_kernel<<<dim3(DFF_ / 32, D_ / 32, 1), 256, 0, stream>>>(
            W2_i, wt, DFF_, D_);
        gemm_bf16_kernel<4><<<gD, 256, 0, stream>>>(h_b, wt, b2_i, tmp_f, nullptr,
                                                    Mtok, D_, DFF_, 0, 0, 0, 0, 0);

        float* dst = (i == NL_ - 1) ? (float*)d_out : x_f;
        add_ln_kernel<<<Mtok, 256, 0, stream>>>(tmp_f, x_f, dst, x_b);
    }
    (void)in_sizes; (void)n_in; (void)out_size; (void)ws_size;
}
